// trajGRUCell_50800873177510
// MI455X (gfx1250) — matrix operand order repaired and checked by reading, not compiled, not run on hardware
//
#include <hip/hip_runtime.h>
#include <hip/hip_bf16.h>
#include <math.h>

// ---------------------------------------------------------------------------
// TrajGRU cell for MI455X (gfx1250, wave32, WMMA).
//
//   K0  pack 1x1-conv weights (update/reset/out _ww concat, N=288, K=1248)
//       into v_wmma_f32_16x16x32_bf16 A-fragment layout (bf16).
//   K1  conv5x5 #1  (concat[x,prev] 104ch -> 32ch) + ReLU   (f32 VALU)
//   K2  conv5x5 #2  (32ch -> 26ch flow fields)     + ReLU   (f32 VALU)
//   K3  trajectory trilinear sample: builds Mf (K=1248 x M=36864) in bf16,
//       written DIRECTLY in WMMA B-fragment layout (no transpose later).
//   K4  big GEMM (M=36864, K=1248, N=288) on v_wmma_f32_16x16x32_bf16,
//       register-blocked 2x2 tiles per wave (16 FLOP/B from cache),
//       + fused epilogue: 3x3 convs of x, sigmoid gates, leaky relu,
//       GRU blend -> d_out.  No 42MB gate tensor ever materialized.
//
// Scratch layout in d_ws (~101 MB):
//   [WgP  bf16  288*1248        =   718,848 B]
//   [MfP  bf16  1248*36864      = 92,012,544 B]   (fits 192MB L2)
//   [o1   f32   4*32*9216       =  4,718,592 B]
//   [o2   f32   4*26*9216       =  3,833,856 B]
// ---------------------------------------------------------------------------

typedef __attribute__((ext_vector_type(16))) __bf16 v16bf;
typedef __attribute__((ext_vector_type(8)))  float  v8f;

#define Bn   4
#define CIN  8
#define CH   96
#define LL   13
#define HH   96
#define WW2  96
#define HWP  9216            // H*W
#define KDIM 1248            // CH*LL
#define NDIM 288             // 3*CH
#define MDIM 36864           // B*HW
#define KT   39              // KDIM/32
#define NT   18              // NDIM/16
#define MT   2304            // MDIM/16
#define FRAG_ELEMS 512       // 32 lanes * 16 bf16 per fragment

// ---- WMMA fragment address maps (CDNA5 ISA 7.12.2, 16-bit A 16x32 / B 32x16)
// A (16x32): lane = half*16 + m,  half = (k>>3)&1,  j = (k&7) + 8*((k>>4)&1)
// B (32x16): lane = half*16 + n,  half = (k>>3)&1,  j = (k&7) + 8*((k>>4)&1)
__device__ __forceinline__ size_t a_frag_off(int ntile, int ktile, int k, int m) {
    int half = (k >> 3) & 1;
    int j    = (k & 7) + (((k >> 4) & 1) << 3);
    int lane = half * 16 + m;
    return (size_t)(ntile * KT + ktile) * FRAG_ELEMS + lane * 16 + j;
}
__device__ __forceinline__ size_t b_frag_off(int mtile, int ktile, int k, int n) {
    int lane = (((k >> 3) & 1) << 4) | n;
    int j    = (k & 7) + (((k >> 4) & 1) << 3);
    return (size_t)(mtile * KT + ktile) * FRAG_ELEMS + lane * 16 + j;
}

// ---------------- K0: pack 1x1 weights (concat update/reset/out) -----------
__global__ void k_pack_w(const float* __restrict__ upd_ww,
                         const float* __restrict__ rst_ww,
                         const float* __restrict__ out_ww,
                         __bf16* __restrict__ WgP) {
    int idx = blockIdx.x * 256 + threadIdx.x;      // NDIM*KDIM = 359424 exact
    int k = idx % KDIM;
    int n = idx / KDIM;
    float v;
    if (n < CH)            v = upd_ww[(size_t)n * KDIM + k];
    else if (n < 2 * CH)   v = rst_ww[(size_t)(n - CH) * KDIM + k];
    else                   v = out_ww[(size_t)(n - 2 * CH) * KDIM + k];
    WgP[a_frag_off(n >> 4, k >> 5, k & 31, n & 15)] = (__bf16)v;
}

// ---------------- K1: conv5x5 (x ++ prev_state) -> o1 (32ch) + relu --------
__global__ void k_conv1(const float* __restrict__ x,
                        const float* __restrict__ prev,
                        const float* __restrict__ w1,
                        const float* __restrict__ b1,
                        float* __restrict__ o1) {
    int idx = blockIdx.x * 256 + threadIdx.x;      // 4*32*9216 = 1179648 exact
    int w  = idx % WW2;
    int h  = (idx / WW2) % HH;
    int oc = (idx / HWP) % 32;
    int b  = idx / (HWP * 32);
    const float* bx = x    + (size_t)b * CIN * HWP;
    const float* bp = prev + (size_t)b * CH  * HWP;
    float acc = b1[oc];
    for (int c = 0; c < CIN + CH; ++c) {
        const float* sp = (c < CIN) ? (bx + (size_t)c * HWP)
                                    : (bp + (size_t)(c - CIN) * HWP);
        const float* wp = w1 + ((size_t)oc * (CIN + CH) + c) * 25;
        #pragma unroll
        for (int ky = 0; ky < 5; ++ky) {
            int yy = h + ky - 2;
            if (yy < 0 || yy >= HH) continue;
            const float* row = sp + yy * WW2;
            #pragma unroll
            for (int kx = 0; kx < 5; ++kx) {
                int xx = w + kx - 2;
                if (xx < 0 || xx >= WW2) continue;
                acc = fmaf(row[xx], wp[ky * 5 + kx], acc);
            }
        }
    }
    o1[idx] = fmaxf(acc, 0.0f);
}

// ---------------- K2: conv5x5 o1 -> o2 (26ch flows) + relu -----------------
__global__ void k_conv2(const float* __restrict__ o1,
                        const float* __restrict__ w2,
                        const float* __restrict__ b2,
                        float* __restrict__ o2) {
    int idx = blockIdx.x * 256 + threadIdx.x;      // 4*26*9216 = 958464 exact
    int w  = idx % WW2;
    int h  = (idx / WW2) % HH;
    int oc = (idx / HWP) % (2 * LL);
    int b  = idx / (HWP * 2 * LL);
    const float* base = o1 + (size_t)b * 32 * HWP;
    float acc = b2[oc];
    for (int c = 0; c < 32; ++c) {
        const float* sp = base + (size_t)c * HWP;
        const float* wp = w2 + ((size_t)oc * 32 + c) * 25;
        #pragma unroll
        for (int ky = 0; ky < 5; ++ky) {
            int yy = h + ky - 2;
            if (yy < 0 || yy >= HH) continue;
            const float* row = sp + yy * WW2;
            #pragma unroll
            for (int kx = 0; kx < 5; ++kx) {
                int xx = w + kx - 2;
                if (xx < 0 || xx >= WW2) continue;
                acc = fmaf(row[xx], wp[ky * 5 + kx], acc);
            }
        }
    }
    o2[idx] = fmaxf(acc, 0.0f);
}

// ------- K3: trilinear trajectory sample -> Mf, packed as WMMA B frags -----
__global__ void k_sample_pack(const float* __restrict__ prev,
                              const float* __restrict__ o2,
                              __bf16* __restrict__ MfP) {
    int idx = blockIdx.x * 256 + threadIdx.x;      // 4*13*9216 = 479232 exact
    int w = idx % WW2;
    int h = (idx / WW2) % HH;
    int l = (idx / HWP) % LL;
    int b = idx / (HWP * LL);

    float gx = o2[((size_t)(b * 2 * LL + l)      ) * HWP + h * WW2 + w];
    float gy = o2[((size_t)(b * 2 * LL + LL + l) ) * HWP + h * WW2 + w];

    float gxn = gx * (2.0f / (WW2 - 1)) - 1.0f;
    float gyn = gy * (2.0f / (HH  - 1)) - 1.0f;
    float ix  = ((gxn + 1.0f) * WW2 - 1.0f) * 0.5f;
    float iy  = ((gyn + 1.0f) * HH  - 1.0f) * 0.5f;
    float gzn = (float)(l + 1) * (2.0f / (LL - 1)) - 1.0f;
    float iz  = ((gzn + 1.0f) * LL - 1.0f) * 0.5f;

    float x0 = floorf(ix), y0 = floorf(iy), z0 = floorf(iz);

    // z weights folded (both z taps hit the same 2D slice of prev_state)
    float wz0 = 1.0f - fabsf(iz - z0);
    float wz1 = 1.0f - fabsf(iz - z0 - 1.0f);
    float mz0 = (z0 >= 0.0f        && z0 <= (float)(LL - 1)) ? 1.0f : 0.0f;
    float mz1 = (z0 + 1.0f >= 0.0f && z0 + 1.0f <= (float)(LL - 1)) ? 1.0f : 0.0f;
    float wzs = wz0 * mz0 + wz1 * mz1;

    float wt[4];
    const float* tap[4];
    int t = 0;
    const float* bprev = prev + (size_t)b * CH * HWP;
    #pragma unroll
    for (int dy = 0; dy < 2; ++dy) {
        float yc = y0 + (float)dy;
        float wy = 1.0f - fabsf(iy - yc);
        float my = (yc >= 0.0f && yc <= (float)(HH - 1)) ? 1.0f : 0.0f;
        int yi = (int)fminf(fmaxf(yc, 0.0f), (float)(HH - 1));
        #pragma unroll
        for (int dx = 0; dx < 2; ++dx) {
            float xc = x0 + (float)dx;
            float wx = 1.0f - fabsf(ix - xc);
            float mx = (xc >= 0.0f && xc <= (float)(WW2 - 1)) ? 1.0f : 0.0f;
            int xi = (int)fminf(fmaxf(xc, 0.0f), (float)(WW2 - 1));
            wt[t]  = wzs * wy * wx * my * mx;
            tap[t] = bprev + yi * WW2 + xi;
            ++t;
        }
    }

    int m     = b * HWP + h * WW2 + w;   // global GEMM column
    int mtile = m >> 4;
    int col   = m & 15;
    for (int c = 0; c < CH; ++c) {
        size_t coff = (size_t)c * HWP;
        float v = wt[0] * tap[0][coff] + wt[1] * tap[1][coff]
                + wt[2] * tap[2][coff] + wt[3] * tap[3][coff];
        int k = c * LL + l;              // matches Mf.reshape(B, Ch*L, H, W)
        MfP[b_frag_off(mtile, k >> 5, k & 31, col)] = (__bf16)v;
    }
}

// ------- K4: WMMA GEMM, 2x2 register-blocked, + fused GRU epilogue ---------
// Block = 288 threads = 9 waves, covers 2 m-tiles (32 pixels).
// Wave w owns n-tiles {2w, 2w+1} x m-tiles {0,1}: 4 v8f accumulators.
// Per k-step: 4 fragment loads (4KB) feed 4 WMMAs (64 KFLOP) = 16 FLOP/B.
__global__ void __launch_bounds__(288)
k_gemm_fuse(const __bf16* __restrict__ WgP,
            const __bf16* __restrict__ MfP,
            const float* __restrict__ x,
            const float* __restrict__ prev,
            const float* __restrict__ upd_w, const float* __restrict__ upd_b,
            const float* __restrict__ rst_w, const float* __restrict__ rst_b,
            const float* __restrict__ out_w, const float* __restrict__ out_b,
            const float* __restrict__ upd_wb,
            const float* __restrict__ rst_wb,
            const float* __restrict__ out_wb,
            float* __restrict__ out) {
    __shared__ float sg[NDIM * 32];          // 288 channels x 32 pixels (36KB)

    const int tid  = threadIdx.x;
    const int wave = tid >> 5;               // 0..8
    const int lane = tid & 31;
    const int mt0  = 2 * blockIdx.x;         // 0..2302 step 2
    const int mt1  = mt0 + 1;
    const int nt0  = 2 * wave;
    const int nt1  = 2 * wave + 1;

    v8f c00 = {}, c01 = {}, c10 = {}, c11 = {};
    const __bf16* bbase0 = MfP + (size_t)mt0 * KT * FRAG_ELEMS + lane * 16;
    const __bf16* bbase1 = MfP + (size_t)mt1 * KT * FRAG_ELEMS + lane * 16;
    const __bf16* abase0 = WgP + (size_t)nt0 * KT * FRAG_ELEMS + lane * 16;
    const __bf16* abase1 = WgP + (size_t)nt1 * KT * FRAG_ELEMS + lane * 16;

    for (int kt = 0; kt < KT; ++kt) {
        if (kt + 1 < KT) {                   // pull next B frags toward L0/L2
            __builtin_prefetch(bbase0 + (size_t)(kt + 1) * FRAG_ELEMS, 0, 3);
            __builtin_prefetch(bbase1 + (size_t)(kt + 1) * FRAG_ELEMS, 0, 3);
        }
        v16bf b0 = *(const v16bf*)(bbase0 + (size_t)kt * FRAG_ELEMS);
        v16bf b1 = *(const v16bf*)(bbase1 + (size_t)kt * FRAG_ELEMS);
        v16bf a0 = *(const v16bf*)(abase0 + (size_t)kt * FRAG_ELEMS);
        v16bf a1 = *(const v16bf*)(abase1 + (size_t)kt * FRAG_ELEMS);
        c00 = __builtin_amdgcn_wmma_f32_16x16x32_bf16(false, a0, false, b0,
                                                      (short)0, c00, false, false);
        c01 = __builtin_amdgcn_wmma_f32_16x16x32_bf16(false, a0, false, b1,
                                                      (short)0, c01, false, false);
        c10 = __builtin_amdgcn_wmma_f32_16x16x32_bf16(false, a1, false, b0,
                                                      (short)0, c10, false, false);
        c11 = __builtin_amdgcn_wmma_f32_16x16x32_bf16(false, a1, false, b1,
                                                      (short)0, c11, false, false);
    }

    // scatter D fragments into LDS: D element (row = r + 8*(lane/16), col = lane%16)
    {
        int half = lane >> 4;
        int col  = lane & 15;
        #pragma unroll
        for (int r = 0; r < 8; ++r) {
            int ch0 = nt0 * 16 + r + 8 * half;
            int ch1 = nt1 * 16 + r + 8 * half;
            sg[ch0 * 32 + col]      = c00[r];
            sg[ch0 * 32 + 16 + col] = c01[r];
            sg[ch1 * 32 + col]      = c10[r];
            sg[ch1 * 32 + 16 + col] = c11[r];
        }
    }
    __syncthreads();

    // fused epilogue: 32 pixels x 96 channels
    for (int i = tid; i < 32 * CH; i += 288) {
        int pix = i & 31;
        int ch  = i >> 5;
        int m   = mt0 * 16 + pix;
        int w   = m % WW2;
        int h   = (m / WW2) % HH;
        int b   = m / HWP;

        // three 3x3 convs of x, sharing the x loads
        float au = 0.f, ar = 0.f, ao = 0.f;
        const float* bx = x + (size_t)b * CIN * HWP;
        for (int cin = 0; cin < CIN; ++cin) {
            const float* sp = bx + (size_t)cin * HWP;
            const float* wub = upd_w + ((size_t)ch * CIN + cin) * 9;
            const float* wrb = rst_w + ((size_t)ch * CIN + cin) * 9;
            const float* wob = out_w + ((size_t)ch * CIN + cin) * 9;
            #pragma unroll
            for (int ky = 0; ky < 3; ++ky) {
                int yy = h + ky - 1;
                if (yy < 0 || yy >= HH) continue;
                const float* row = sp + yy * WW2;
                #pragma unroll
                for (int kx = 0; kx < 3; ++kx) {
                    int xx = w + kx - 1;
                    if (xx < 0 || xx >= WW2) continue;
                    float xv = row[xx];
                    int wi = ky * 3 + kx;
                    au = fmaf(xv, wub[wi], au);
                    ar = fmaf(xv, wrb[wi], ar);
                    ao = fmaf(xv, wob[wi], ao);
                }
            }
        }

        float gu = sg[(ch)          * 32 + pix] + upd_wb[ch];
        float gr = sg[(CH + ch)     * 32 + pix] + rst_wb[ch];
        float go = sg[(2 * CH + ch) * 32 + pix] + out_wb[ch];

        float u  = 1.0f / (1.0f + __expf(-(au + upd_b[ch] + gu)));
        float r  = 1.0f / (1.0f + __expf(-(ar + rst_b[ch] + gr)));
        float v  = ao + out_b[ch] + r * go;
        float lv = (v >= 0.0f) ? v : 0.2f * v;

        size_t oi = (((size_t)b * CH + ch) * HH + h) * WW2 + w;
        out[oi] = prev[oi] * u + lv * (1.0f - u);
    }
}

// ---------------------------------------------------------------------------
extern "C" void kernel_launch(void* const* d_in, const int* in_sizes, int n_in,
                              void* d_out, int out_size, void* d_ws, size_t ws_size,
                              hipStream_t stream) {
    const float* x       = (const float*)d_in[0];
    const float* prev    = (const float*)d_in[1];
    const float* sub_w1  = (const float*)d_in[2];
    const float* sub_b1  = (const float*)d_in[3];
    const float* sub_w2  = (const float*)d_in[4];
    const float* sub_b2  = (const float*)d_in[5];
    const float* rst_w   = (const float*)d_in[6];
    const float* rst_b   = (const float*)d_in[7];
    const float* upd_w   = (const float*)d_in[8];
    const float* upd_b   = (const float*)d_in[9];
    const float* out_w   = (const float*)d_in[10];
    const float* out_b   = (const float*)d_in[11];
    const float* rst_ww  = (const float*)d_in[12];
    const float* rst_wb  = (const float*)d_in[13];
    const float* upd_ww  = (const float*)d_in[14];
    const float* upd_wb  = (const float*)d_in[15];
    const float* out_ww  = (const float*)d_in[16];
    const float* out_wb  = (const float*)d_in[17];
    float* out = (float*)d_out;

    char* ws = (char*)d_ws;
    const size_t WG_BYTES = (size_t)NDIM * KDIM * 2;            //    718,848
    const size_t MF_BYTES = (size_t)KDIM * MDIM * 2;            // 92,012,544
    const size_t O1_BYTES = (size_t)Bn * 32 * HWP * 4;          //  4,718,592
    __bf16* WgP = (__bf16*)(ws);
    __bf16* MfP = (__bf16*)(ws + WG_BYTES);
    float*  o1  = (float*)(ws + WG_BYTES + MF_BYTES);
    float*  o2  = (float*)(ws + WG_BYTES + MF_BYTES + O1_BYTES);

    k_pack_w<<<(NDIM * KDIM) / 256, 256, 0, stream>>>(upd_ww, rst_ww, out_ww, WgP);
    k_conv1<<<(Bn * 32 * HWP) / 256, 256, 0, stream>>>(x, prev, sub_w1, sub_b1, o1);
    k_conv2<<<(Bn * 2 * LL * HWP) / 256, 256, 0, stream>>>(o1, sub_w2, sub_b2, o2);
    k_sample_pack<<<(Bn * LL * HWP) / 256, 256, 0, stream>>>(prev, o2, MfP);
    k_gemm_fuse<<<MT / 2, 288, 0, stream>>>(WgP, MfP, x, prev,
                                            upd_w, upd_b, rst_w, rst_b, out_w, out_b,
                                            upd_wb, rst_wb, out_wb, out);
}
